// NSAEncoder_83159156785780
// MI455X (gfx1250) — matrix operand order repaired and checked by reading, not compiled, not run on hardware
//
#include <hip/hip_runtime.h>
#include <hip/hip_bf16.h>
#include <math.h>

// ---------------------------------------------------------------------------
// Problem constants (match reference)
// ---------------------------------------------------------------------------
#define BATCH   8
#define NNODE   1024
#define DIN     128
#define DH      256          // model dim
#define HEADS   8
#define HDIM    32           // head dim == WMMA K for bf16 (16x16x32)
#define FFDIM   1024         // 4 * DH
#define ROWS    (BATCH * NNODE)   // 8192
#define NEGBIAS (-1e9f)

typedef __attribute__((ext_vector_type(16))) __bf16    bf16x16;
typedef __attribute__((ext_vector_type(8)))  float     f32x8;
typedef __attribute__((ext_vector_type(4)))  unsigned  v4u;
typedef __attribute__((ext_vector_type(8)))  int       v8i_;
typedef __attribute__((ext_vector_type(4)))  int       v4i_;

#if __has_builtin(__builtin_amdgcn_tensor_load_to_lds) && \
    __has_builtin(__builtin_amdgcn_s_wait_tensorcnt)
#define HAVE_TDM 1
#endif

// pack two floats into two bf16 in one u32 (lo = a, hi = b)
__device__ __forceinline__ unsigned pk2(float a, float b) {
    union { __bf16 h[2]; unsigned u; } x;
    x.h[0] = (__bf16)a;
    x.h[1] = (__bf16)b;
    return x.u;
}

// Branch-free tanh: native gfx1250 v_tanh_f32 when available.
__device__ __forceinline__ float fast_tanh(float u) {
#if __has_builtin(__builtin_amdgcn_tanhf)
    return __builtin_amdgcn_tanhf(u);
#else
    float a = fabsf(u);
    float t = __expf(2.0f * a);
    float y = 1.0f - 2.0f / (t + 1.0f);
    return copysignf(y, u);
#endif
}

// ---------------------------------------------------------------------------
// Helpers: adjacency mask build, f32 -> bf16 conversion, weight pre-swizzle
// ---------------------------------------------------------------------------
__global__ void zero_u32_kernel(unsigned* p, int n) {
    int i = blockIdx.x * blockDim.x + threadIdx.x;
    if (i < n) p[i] = 0u;
}

__global__ void scatter_edges_kernel(const int* __restrict__ ei,
                                     unsigned* __restrict__ mask, int E) {
    int e = blockIdx.x * blockDim.x + threadIdx.x;
    if (e >= E) return;
    int src = ei[e];
    int dst = ei[E + e];
    int b = src >> 10;
    int i = src & (NNODE - 1);
    int j = dst & (NNODE - 1);
    atomicOr(&mask[((size_t)(b * NNODE + i)) * (NNODE / 32) + (j >> 5)],
             1u << (j & 31));
}

// paired f32 -> bf16 (n must be even)
__global__ void f32_to_bf16_kernel(const float* __restrict__ in,
                                   unsigned* __restrict__ out, int npairs) {
    int i = blockIdx.x * blockDim.x + threadIdx.x;
    if (i < npairs) out[i] = pk2(in[2 * i], in[2 * i + 1]);
}

// Pre-swizzle a weight W[K,Nc] (f32 row-major) into bf16 fragment order:
//   tile (nb over Nc/128, kb over K/32) -> [c:8][half:2][n:16][e:16]
// so a GEMM weight tile is one contiguous 8KB block in WMMA B-fragment order.
// B-fragment mapping: k (0..31) -> half=k>>4, e=k&15 ; col -> c=col>>4, n=col&15
__global__ void weight_swizzle_kernel(const float* __restrict__ W,
                                      unsigned* __restrict__ out,
                                      int K, int Nc) {
    int id = blockIdx.x * blockDim.x + threadIdx.x;   // u32 (bf16-pair) index
    int total = (K * Nc) >> 1;
    if (id >= total) return;
    int ep = id & 7;                 // element pair (e = 2*ep, 2*ep+1)
    int n  = (id >> 3) & 15;
    int ch = (id >> 7) & 15;
    int c  = ch >> 1, hf = ch & 1;
    int rest = id >> 11;
    int ksteps = K >> 5;
    int kb = rest % ksteps;
    int nb = rest / ksteps;
    int k   = kb * 32 + hf * 16 + ep * 2;
    int col = nb * 128 + c * 16 + n;
    out[id] = pk2(W[(size_t)k * Nc + col], W[(size_t)(k + 1) * Nc + col]);
}

// ---------------------------------------------------------------------------
// Tiled WMMA GEMM:  act(A[M,K](bf16) @ Wsw(pre-swizzled bf16) + bias)
// 64x128 tile, BK=32, 128 threads (4 waves); wave w owns a 16x128 strip.
// A tile: per-thread 16B chunks land contiguously in fragment order.
// W tile: one contiguous 8KB block -> TDM tensor_load_to_lds (wave 0) or
//         cooperative b128 copy fallback.
// Outputs: Cf (f32, optional), Cb (bf16, optional; vt=1 stores head-transposed
//          V layout [b,h,32,1024] for the attention kernel).
// ---------------------------------------------------------------------------
__global__ __launch_bounds__(128) void gemm_bias_act_kernel(
    const __bf16* __restrict__ A, const __bf16* __restrict__ Wsw,
    const float* __restrict__ bias, float* __restrict__ Cf,
    __bf16* __restrict__ Cb, int M, int K, int Nc, int act, int vt)
{
    __shared__ bf16x16 Afrag[4][2][16];   // [row-strip][half][row%16]   4 KB
    __shared__ bf16x16 Bfrag[8][2][16];   // [col-tile ][half][col%16]   8 KB

    const int t    = threadIdx.x;
    const int lane = t & 31;
    const int wv   = t >> 5;
    const int half = lane >> 4;
    const int lr   = lane & 15;
    const int m0   = blockIdx.x * 64;
    const int ksteps = K >> 5;

    f32x8 acc[8] = {};

    for (int ks = 0; ks < ksteps; ++ks) {
        const int k0 = ks * 32;
        // ---- A tile 64x32 bf16: 256 x 16B chunks, 2 per thread.
        // chunk q: r=q>>2, hf=(q>>1)&1, g=q&1 covers k in [hf*16+g*8, +8)
        // -> fragment [r>>4][hf][r&15] elements [g*8, g*8+8)  (contiguous)
        #pragma unroll
        for (int i = 0; i < 2; ++i) {
            int q  = t + i * 128;
            int r  = q >> 2;
            int hf = (q >> 1) & 1;
            int g  = q & 1;
            const uint4 v =
                *(const uint4*)(A + (size_t)(m0 + r) * K + k0 + hf * 16 + g * 8);
            *(uint4*)((unsigned short*)&Afrag[r >> 4][hf][r & 15] + g * 8) = v;
        }
        // ---- W tile: contiguous 8KB in fragment order ----
        const __bf16* wt =
            Wsw + ((size_t)blockIdx.y * ksteps + ks) * 4096;
#ifdef HAVE_TDM
        if (wv == 0) {
            unsigned lds_off = (unsigned)(size_t)(void*)&Bfrag[0][0][0];
            unsigned long long ga = (unsigned long long)(const void*)wt;
            v4u g0;
            g0[0] = 1u;                                   // count=1, user mode
            g0[1] = lds_off;                              // lds_addr (bytes)
            g0[2] = (unsigned)ga;                         // global_addr[31:0]
            g0[3] = (unsigned)((ga >> 32) & 0x01FFFFFFull) | (2u << 30); // type=2
            v8i_ g1 = {};
            g1[0] = (1 << 16);            // data_size = 1 -> 2 bytes
            g1[1] = (int)(4096u << 16);   // tensor_dim0[15:0] = 4096
            g1[2] = (int)(1u << 16);      // tensor_dim0 hi | tensor_dim1 = 1
            g1[3] = (int)(4096u << 16);   // tensor_dim1 hi | tile_dim0 = 4096
            g1[4] = 1;                    // tile_dim1 = 1
            g1[5] = 4096;                 // tensor_dim0_stride low32
            v4i_ g2 = {};
            v4i_ g3 = {};
            v8i_ g4 = {};                 // unused (count=1)
            __builtin_amdgcn_tensor_load_to_lds(g0, g1, g2, g3, g4, 0);
            __builtin_amdgcn_s_wait_tensorcnt(0);
        }
#else
        #pragma unroll
        for (int i = 0; i < 4; ++i) {
            int q = t + i * 128;
            const uint4 v = *(const uint4*)((const unsigned short*)wt + q * 8);
            *(uint4*)((unsigned short*)&Bfrag[0][0][0] + q * 8) = v;
        }
#endif
        __syncthreads();

        const bf16x16 af = Afrag[wv][half][lr];
        bf16x16 bfr[8];
        #pragma unroll
        for (int c = 0; c < 8; ++c) bfr[c] = Bfrag[c][half][lr];
        #pragma unroll
        for (int c = 0; c < 8; ++c) {
            acc[c] = __builtin_amdgcn_wmma_f32_16x16x32_bf16(
                false, af, false, bfr[c], (short)0, acc[c], false, false);
        }
        __syncthreads();
    }

    // Epilogue: C-fragment row = 16*wv + 8*half + r, col = 16*c + lr
    #pragma unroll
    for (int c = 0; c < 8; ++c) {
        #pragma unroll
        for (int r = 0; r < 8; ++r) {
            int row = m0 + wv * 16 + half * 8 + r;
            int col = blockIdx.y * 128 + c * 16 + lr;
            float x = acc[c][r] + bias[col];
            if (act == 1) {
                float x3 = x * x * x;
                float u  = 0.7978845608028654f * (x + 0.044715f * x3);
                x = 0.5f * x * (1.0f + fast_tanh(u));
            }
            if (Cf) Cf[(size_t)row * Nc + col] = x;
            if (Cb) {
                if (vt) {   // head-transposed V: [b][head][cc(32)][key(1024)]
                    int bb = row >> 10, key = row & (NNODE - 1);
                    int head = col >> 5, cc = col & 31;
                    Cb[(((size_t)bb * HEADS + head) * HDIM + cc) * NNODE + key] =
                        (__bf16)x;
                } else {
                    Cb[(size_t)row * Nc + col] = (__bf16)x;
                }
            }
        }
    }
}

// ---------------------------------------------------------------------------
// Flash attention: one wave owns a 16-query strip of one (batch, head).
// Q,K bf16 row-major; V bf16 head-transposed [b,h,32,1024]; O bf16 row-major.
// ---------------------------------------------------------------------------
__global__ __launch_bounds__(128) void flash_attn_kernel(
    const __bf16* __restrict__ Q, const __bf16* __restrict__ Kb,
    const __bf16* __restrict__ Vt, __bf16* __restrict__ O,
    const unsigned* __restrict__ mask, int use_mask)
{
    __shared__ bf16x16 Ps[4][2][16];   // per-wave P staging [wave][half][row]

    const int t    = threadIdx.x;
    const int lane = t & 31;
    const int wv   = t >> 5;
    const int wid  = blockIdx.x * 4 + wv;
    const int qt   = wid & 63;
    const int hd   = (wid >> 6) & 7;
    const int b    = wid >> 9;
    const int half = lane >> 4;
    const int lr   = lane & 15;
    const float scale = 0.17677669529663687f;   // 1/sqrt(32)

    // ---- Q fragment: two 16B loads (A-frag k runs are contiguous) ----
    union Frag { bf16x16 v; uint4 u[2]; };
    Frag qa;
    {
        const __bf16* qrow =
            Q + ((size_t)(b * NNODE + qt * 16 + lr)) * DH + hd * HDIM;
        qa.u[0] = *(const uint4*)(qrow + half * 16);
        qa.u[1] = *(const uint4*)(qrow + half * 16 + 8);
    }
    const bf16x16 qf = qa.v;

    // V base pointers (head-transposed rows for hd-cols lr and 16+lr)
    const __bf16* vrow0 =
        Vt + (((size_t)b * HEADS + hd) * HDIM + lr) * NNODE;
    const __bf16* vrow1 =
        Vt + (((size_t)b * HEADS + hd) * HDIM + 16 + lr) * NNODE;

    float mrow[8], lrow[8];
    #pragma unroll
    for (int r = 0; r < 8; ++r) { mrow[r] = -1e30f; lrow[r] = 0.0f; }
    f32x8 o0 = {}, o1 = {};

    for (int j0 = 0; j0 < NNODE; j0 += 32) {
        // ---- scores for two 16-key sub-tiles ----
        f32x8 s[2];
        #pragma unroll
        for (int jj = 0; jj < 2; ++jj) {
            int key = j0 + jj * 16 + lr;
            const bf16x16 kf = *(const bf16x16*)(
                Kb + ((size_t)(b * NNODE + key)) * DH + hd * HDIM + half * 16);
            f32x8 z = {};
            s[jj] = __builtin_amdgcn_wmma_f32_16x16x32_bf16(
                false, qf, false, kf, (short)0, z, false, false);
        }
        // ---- scale + adjacency bias ----
        #pragma unroll
        for (int r = 0; r < 8; ++r) {
            unsigned wbits = 0xFFFFFFFFu;
            if (use_mask) {
                int row = qt * 16 + half * 8 + r;
                wbits = mask[((size_t)(b * NNODE + row)) * (NNODE / 32) +
                             (j0 >> 5)];
            }
            #pragma unroll
            for (int jj = 0; jj < 2; ++jj) {
                float val = s[jj][r] * scale;
                if (!((wbits >> (jj * 16 + lr)) & 1u)) val += NEGBIAS;
                s[jj][r] = val;
            }
        }
        // ---- online softmax over the 32 keys ----
        const int hlf2 = (lr >> 3) & 1;
        #pragma unroll
        for (int r = 0; r < 8; ++r) {
            float tmax = fmaxf(s[0][r], s[1][r]);
            #pragma unroll
            for (int off = 8; off >= 1; off >>= 1)
                tmax = fmaxf(tmax, __shfl_xor(tmax, off, 32));
            float mnew  = fmaxf(mrow[r], tmax);
            float alpha = __expf(mrow[r] - mnew);
            float p0 = __expf(s[0][r] - mnew);
            float p1 = __expf(s[1][r] - mnew);
            float psum = p0 + p1;
            #pragma unroll
            for (int off = 8; off >= 1; off >>= 1)
                psum += __shfl_xor(psum, off, 32);
            lrow[r] = lrow[r] * alpha + psum;
            mrow[r] = mnew;
            o0[r] *= alpha;
            o1[r] *= alpha;
            unsigned short* pr =
                (unsigned short*)&Ps[wv][0][half * 8 + r];
            union { __bf16 h; unsigned short u; } c0, c1;
            c0.h = (__bf16)p0; c1.h = (__bf16)p1;
            pr[lr]       = c0.u;
            pr[256 + lr] = c1.u;
        }
        __syncthreads();

        // ---- O += P(16x32) @ V(32keys x 32hd): two 32B V loads, two WMMAs
        const bf16x16 pf  = Ps[wv][half][lr];
        const bf16x16 vf0 = *(const bf16x16*)(vrow0 + j0 + half * 16);
        const bf16x16 vf1 = *(const bf16x16*)(vrow1 + j0 + half * 16);
        o0 = __builtin_amdgcn_wmma_f32_16x16x32_bf16(
            false, pf, false, vf0, (short)0, o0, false, false);
        o1 = __builtin_amdgcn_wmma_f32_16x16x32_bf16(
            false, pf, false, vf1, (short)0, o1, false, false);
        __syncthreads();
    }

    #pragma unroll
    for (int r = 0; r < 8; ++r) {
        float inv = 1.0f / lrow[r];
        int row = qt * 16 + half * 8 + r;
        __bf16* orow = O + ((size_t)(b * NNODE + row)) * DH + hd * HDIM;
        orow[lr]      = (__bf16)(o0[r] * inv);
        orow[16 + lr] = (__bf16)(o1[r] * inv);
    }
}

// ---------------------------------------------------------------------------
// Residual + LayerNorm: hf32 = LN(res + delta)*g + b ; also mirrors to bf16.
// ---------------------------------------------------------------------------
__global__ __launch_bounds__(256) void resid_ln_kernel(
    const float* __restrict__ res, const float* __restrict__ delta,
    const float* __restrict__ g, const float* __restrict__ be,
    float* __restrict__ outf, __bf16* __restrict__ outb)
{
    __shared__ float sums[8], sqs[8];
    const int row  = blockIdx.x;
    const int t    = threadIdx.x;
    const int lane = t & 31;
    const int wv   = t >> 5;

    float v = res[(size_t)row * DH + t] + delta[(size_t)row * DH + t];
    float s1 = v, s2 = v * v;
    #pragma unroll
    for (int off = 16; off >= 1; off >>= 1) {
        s1 += __shfl_xor(s1, off, 32);
        s2 += __shfl_xor(s2, off, 32);
    }
    if (lane == 0) { sums[wv] = s1; sqs[wv] = s2; }
    __syncthreads();
    float tot = 0.0f, tot2 = 0.0f;
    #pragma unroll
    for (int i = 0; i < 8; ++i) { tot += sums[i]; tot2 += sqs[i]; }
    float mu  = tot * (1.0f / DH);
    float var = tot2 * (1.0f / DH) - mu * mu;
    float inv = rsqrtf(var + 1e-5f);
    float y = (v - mu) * inv * g[t] + be[t];
    outf[(size_t)row * DH + t] = y;
    outb[(size_t)row * DH + t] = (__bf16)y;
}

// ---------------------------------------------------------------------------
// Host orchestration
// ---------------------------------------------------------------------------
extern "C" void kernel_launch(void* const* d_in, const int* in_sizes, int n_in,
                              void* d_out, int out_size, void* d_ws, size_t ws_size,
                              hipStream_t stream) {
    (void)n_in; (void)out_size; (void)ws_size;

    const float* x    = (const float*)d_in[0];
    const int*   ei   = (const int*)d_in[1];
    const float* in_b = (const float*)d_in[4];
    // Per-layer params: wq bq wk bk wv bv wo bo w1 b1 w2 b2 ln1g ln1b ln2g ln2b
    auto P = [&](int L, int k) { return (const float*)d_in[5 + L * 16 + k]; };

    const int E = in_sizes[1] / 2;

    // ---- workspace carve-up (bytes) ----
    char* ws = (char*)d_ws;
    auto carve = [&](size_t bytes) { char* p = ws; ws += (bytes + 255) & ~255ull; return p; };
    float*    h     = (float*)   carve((size_t)ROWS * DH * 4);
    float*    delta = (float*)   carve((size_t)ROWS * DH * 4);
    __bf16*   hb    = (__bf16*)  carve((size_t)ROWS * DH * 2);
    __bf16*   qb    = (__bf16*)  carve((size_t)ROWS * DH * 2);
    __bf16*   kb    = (__bf16*)  carve((size_t)ROWS * DH * 2);
    __bf16*   vtb   = (__bf16*)  carve((size_t)ROWS * DH * 2);
    __bf16*   ob    = (__bf16*)  carve((size_t)ROWS * DH * 2);
    __bf16*   ffb   = (__bf16*)  carve((size_t)ROWS * FFDIM * 2);
    __bf16*   xb    = (__bf16*)  carve((size_t)ROWS * DIN * 2);
    unsigned* mask  = (unsigned*)carve((size_t)BATCH * NNODE * (NNODE / 32) * 4);
    __bf16*   wswb  = (__bf16*)  carve((32768ull + 4ull * 786432ull) * 2);

    const int mask_words = BATCH * NNODE * (NNODE / 32);

    // ---- adjacency mask ----
    zero_u32_kernel<<<(mask_words + 255) / 256, 256, 0, stream>>>(mask, mask_words);
    scatter_edges_kernel<<<(E + 255) / 256, 256, 0, stream>>>(ei, mask, E);

    // ---- x -> bf16 ----
    {
        int np = ROWS * DIN / 2;
        f32_to_bf16_kernel<<<(np + 255) / 256, 256, 0, stream>>>(
            x, (unsigned*)xb, np);
    }

    // ---- weight pre-swizzle (once per weight) ----
    size_t woff = 0;
    auto swz = [&](const float* W, int K, int Nc) {
        __bf16* dst = wswb + woff;
        woff += (size_t)K * Nc;
        int total = K * Nc / 2;
        weight_swizzle_kernel<<<(total + 255) / 256, 256, 0, stream>>>(
            W, (unsigned*)dst, K, Nc);
        return dst;
    };
    __bf16* w_in = swz((const float*)d_in[3], DIN, DH);
    __bf16 *w_q[4], *w_k[4], *w_v[4], *w_o[4], *w_1[4], *w_2[4];
    for (int L = 0; L < 4; ++L) {
        w_q[L] = swz(P(L, 0),  DH, DH);
        w_k[L] = swz(P(L, 2),  DH, DH);
        w_v[L] = swz(P(L, 4),  DH, DH);
        w_o[L] = swz(P(L, 6),  DH, DH);
        w_1[L] = swz(P(L, 8),  DH, FFDIM);
        w_2[L] = swz(P(L, 10), FFDIM, DH);
    }

    // ---- input projection: h (f32) + hb (bf16) ----
    gemm_bias_act_kernel<<<dim3(ROWS / 64, DH / 128), 128, 0, stream>>>(
        xb, w_in, in_b, h, hb, ROWS, DIN, DH, 0, 0);

    for (int L = 0; L < 4; ++L) {
        const int use_mask = (L == 0 || L == 2);   // PATTERN = M S M S

        gemm_bias_act_kernel<<<dim3(ROWS / 64, DH / 128), 128, 0, stream>>>(
            hb, w_q[L], P(L, 1), nullptr, qb, ROWS, DH, DH, 0, 0);
        gemm_bias_act_kernel<<<dim3(ROWS / 64, DH / 128), 128, 0, stream>>>(
            hb, w_k[L], P(L, 3), nullptr, kb, ROWS, DH, DH, 0, 0);
        gemm_bias_act_kernel<<<dim3(ROWS / 64, DH / 128), 128, 0, stream>>>(
            hb, w_v[L], P(L, 5), nullptr, vtb, ROWS, DH, DH, 0, 1);

        flash_attn_kernel<<<BATCH * HEADS * (NNODE / 16) / 4, 128, 0, stream>>>(
            qb, kb, vtb, ob, mask, use_mask);

        gemm_bias_act_kernel<<<dim3(ROWS / 64, DH / 128), 128, 0, stream>>>(
            ob, w_o[L], P(L, 7), delta, nullptr, ROWS, DH, DH, 0, 0);
        resid_ln_kernel<<<ROWS, 256, 0, stream>>>(
            h, delta, P(L, 12), P(L, 13), h, hb);

        gemm_bias_act_kernel<<<dim3(ROWS / 64, FFDIM / 128), 128, 0, stream>>>(
            hb, w_1[L], P(L, 9), nullptr, ffb, ROWS, DH, FFDIM, 1, 0);
        gemm_bias_act_kernel<<<dim3(ROWS / 64, DH / 128), 128, 0, stream>>>(
            ffb, w_2[L], P(L, 11), delta, nullptr, ROWS, FFDIM, DH, 0, 0);
        resid_ln_kernel<<<ROWS, 256, 0, stream>>>(
            h, delta, P(L, 14), P(L, 15), h, hb);
    }

    (void)hipMemcpyAsync(d_out, h, (size_t)ROWS * DH * sizeof(float),
                         hipMemcpyDeviceToDevice, stream);
}
